// Qwen3AttentionTP_57011395887254
// MI455X (gfx1250) — matrix operand order repaired and checked by reading, not compiled, not run on hardware
//
#include <hip/hip_runtime.h>

#define DEVINL __device__ __forceinline__

typedef __bf16 bf16;
typedef __attribute__((ext_vector_type(16))) __bf16 v16bf;
typedef __attribute__((ext_vector_type(8)))  __bf16 v8bf;
typedef __attribute__((ext_vector_type(8)))  float  v8f;
typedef __attribute__((ext_vector_type(4)))  unsigned int v4u;
typedef __attribute__((ext_vector_type(8)))  int v8i;
typedef __attribute__((ext_vector_type(4)))  int v4i;

static constexpr int SEQ    = 2048;
static constexpr int HIDDEN = 4096;
static constexpr int NHEADS = 32;
static constexpr int NKVH   = 8;
static constexpr int HD     = 128;

struct BF16x16 { uint4 lo, hi; };

DEVINL v8f vzero() { v8f z = {0.f,0.f,0.f,0.f,0.f,0.f,0.f,0.f}; return z; }

// ---- WMMA bf16: D = A(16x32) * B(32x16) + C, fp32 accum (wave32) ----
DEVINL v8f wmma_bf16(v16bf a, v16bf b, v8f c) {
  return __builtin_amdgcn_wmma_f32_16x16x32_bf16(
      /*neg_a=*/false, a, /*neg_b=*/false, b,
      /*c_mod=*/(short)0, c, /*reuse_a=*/false, /*reuse_b=*/false);
}

// A fragment (16x32, row-major src in LDS, ld in elements).
DEVINL v16bf load_a_frag(const bf16* base, int ld, int lane) {
  const int m  = lane & 15;
  const int kh = (lane >> 4) << 3;            // 0 or 8
  const bf16* p = base + m * ld + kh;
  BF16x16 r;
  r.lo = *(const uint4*)(p);                  // K = kh .. kh+7
  r.hi = *(const uint4*)(p + 16);             // K = kh+16 .. kh+23
  return __builtin_bit_cast(v16bf, r);
}

// B fragment (32x16): column N = lane&15 is a contiguous row of the row-major source.
DEVINL v16bf load_b_frag(const bf16* base, int ld, int lane) {
  const int n  = lane & 15;
  const int k0 = (lane >> 4) << 3;            // 0 or 8
  const bf16* p = base + n * ld + k0;
  BF16x16 r;
  r.lo = *(const uint4*)(p);
  r.hi = *(const uint4*)(p + 16);
  return __builtin_bit_cast(v16bf, r);
}

// ---- Tensor Data Mover: 2D bf16 tile Global -> LDS with optional row padding ----
// D# layout per CDNA5 ISA §8.3/8.4.  pad codes: interval 2<<i DWORDs, amount (code+1) DWORDs.
DEVINL void tdm_load_2d_bf16(unsigned lds_addr, const void* gaddr,
                             unsigned tile_d0, unsigned tile_d1,
                             unsigned tensor_d0, unsigned tensor_d1,
                             unsigned stride_d0,
                             unsigned pad_interval_code, unsigned pad_amount_code,
                             bool pad_en)
{
  unsigned long long ga = (unsigned long long)gaddr;
  v4u g0;
  g0[0] = 1u;                                            // count=1, user descriptor
  g0[1] = lds_addr;                                      // LDS byte offset
  g0[2] = (unsigned)(ga & 0xFFFFFFFFu);                  // global_addr lo
  g0[3] = (unsigned)((ga >> 32) & 0x01FFFFFFu) | 0x80000000u;  // addr hi | type=2
  unsigned dw0 = (1u << 16)                              // data_size = 1 (2 bytes)
               | (pad_en ? (1u << 20) : 0u)
               | (pad_interval_code << 22)
               | (pad_amount_code << 25);
  v8i g1;
  g1[0] = (int)dw0;
  g1[1] = (int)((tensor_d0 & 0xFFFFu) << 16);            // tensor_dim0[15:0]
  g1[2] = (int)(((tensor_d0 >> 16) & 0xFFFFu) | ((tensor_d1 & 0xFFFFu) << 16));
  g1[3] = (int)(((tensor_d1 >> 16) & 0xFFFFu) | ((tile_d0 & 0xFFFFu) << 16));
  g1[4] = (int)(tile_d1 & 0xFFFFu);                      // tile_dim1 ; tile_dim2=0
  g1[5] = (int)stride_d0;                                // tensor_dim0_stride[31:0]
  g1[6] = 0;                                             // stride hi / dim1_stride lo
  g1[7] = 0;
  v4i z4 = {0, 0, 0, 0};
  v8i z8 = {0, 0, 0, 0, 0, 0, 0, 0};
  // amdgpu-toolchain (clang-23) 6-arg form
  __builtin_amdgcn_tensor_load_to_lds(g0, g1, z4, z4, z8, 0);
}

DEVINL unsigned lds_off(const void* p) { return (unsigned)(unsigned long long)p; }

// -------------------- fp32 -> bf16 convert --------------------
struct BF4 { bf16 v[4]; };
__global__ void __launch_bounds__(256)
cvt_f32_bf16(const float* __restrict__ x, bf16* __restrict__ y, int n4) {
  int i = blockIdx.x * blockDim.x + threadIdx.x;
  if (i < n4) {
    float4 v = ((const float4*)x)[i];
    BF4 o;
    o.v[0] = (bf16)v.x; o.v[1] = (bf16)v.y; o.v[2] = (bf16)v.z; o.v[3] = (bf16)v.w;
    ((BF4*)y)[i] = o;
  }
}

// -------------------- GEMM: C[64x128] = A(bf16) * W^T(fp32->bf16) --------------------
// MODE 0: store fp32.  MODE 1/2: Q/K epilogue (bias+rmsnorm+rope).  MODE 3: V epilogue.
template <int MODE>
__global__ void __launch_bounds__(128, 1)
gemm_kernel(const bf16*  __restrict__ A,     // [SEQ][HIDDEN] bf16
            const float* __restrict__ W,     // [Nout][HIDDEN] fp32
            const float* __restrict__ bias,
            const float* __restrict__ normw, // [128]
            const float* __restrict__ cosT,  // [SEQ][128]
            const float* __restrict__ sinT,
            bf16*  __restrict__ outB,        // [heads][SEQ][128]
            float* __restrict__ outF)        // [SEQ][HIDDEN]
{
  constexpr int LDA = 40;                    // 32 + 8 pad (16 DW + 4 DW pad rows)
  __shared__ bf16 As[64 * LDA];
  __shared__ bf16 Bs[128 * LDA];

  const int tid   = threadIdx.x;             // 128 threads = 4 waves
  const int lane  = tid & 31;
  const int wv    = tid >> 5;
  const int sBase = blockIdx.x * 64;
  const int nBase = blockIdx.y * 128;

  v8f acc[8];
  #pragma unroll
  for (int j = 0; j < 8; ++j) acc[j] = vzero();

  const unsigned asOff = lds_off(&As[0]);

  for (int kt = 0; kt < HIDDEN / 32; ++kt) {
    const int k0 = kt * 32;
    // A tile 64x32 bf16 via Tensor Data Mover (pad 16DW rows by 4DW -> LDA=40)
    if (wv == 0) {
      tdm_load_2d_bf16(asOff, &A[(size_t)sBase * HIDDEN + k0],
                       /*tile*/32, 64, /*tensor*/HIDDEN, SEQ,
                       /*stride*/HIDDEN, /*pad_int 16DW*/3, /*pad_amt 4DW*/3, true);
    }
    // B tile 128x32: fp32 -> bf16 by all threads (overlaps the TDM DMA)
    {
      const float* wp = &W[(size_t)(nBase + tid) * HIDDEN + k0];
      if (kt + 1 < HIDDEN / 32) __builtin_prefetch(wp + 32, 0, 1);
      #pragma unroll
      for (int q = 0; q < 4; ++q) {
        float4 f0 = *(const float4*)(wp + q * 8);
        float4 f1 = *(const float4*)(wp + q * 8 + 4);
        v8bf t;
        t[0] = (bf16)f0.x; t[1] = (bf16)f0.y; t[2] = (bf16)f0.z; t[3] = (bf16)f0.w;
        t[4] = (bf16)f1.x; t[5] = (bf16)f1.y; t[6] = (bf16)f1.z; t[7] = (bf16)f1.w;
        *(v8bf*)&Bs[tid * LDA + q * 8] = t;
      }
    }
    if (wv == 0) __builtin_amdgcn_s_wait_tensorcnt(0);
    __syncthreads();
    v16bf af = load_a_frag(&As[(wv * 16) * LDA], LDA, lane);
    #pragma unroll
    for (int j = 0; j < 8; ++j) {
      v16bf bfr = load_b_frag(&Bs[(j * 16) * LDA], LDA, lane);
      acc[j] = wmma_bf16(af, bfr, acc[j]);
    }
    __syncthreads();
  }

  // ---- epilogue (C layout: lane&15 = column, rows = (lane>>4)*8 + r) ----
  const int nloc = lane & 15;
  const int rhi  = (lane >> 4) << 3;
  const int rowb = sBase + wv * 16 + rhi;

  if constexpr (MODE == 0) {
    #pragma unroll
    for (int j = 0; j < 8; ++j)
      #pragma unroll
      for (int r = 0; r < 8; ++r)
        outF[(size_t)(rowb + r) * HIDDEN + nBase + j * 16 + nloc] = acc[j][r];
    return;
  } else {
    float x[8][8];
    #pragma unroll
    for (int j = 0; j < 8; ++j) {
      float b = bias ? bias[nBase + j * 16 + nloc] : 0.f;
      #pragma unroll
      for (int r = 0; r < 8; ++r) x[j][r] = acc[j][r] + b;
    }
    const int head = blockIdx.y;
    bf16* outH = outB + (size_t)head * SEQ * HD;

    if constexpr (MODE == 3) {
      #pragma unroll
      for (int j = 0; j < 8; ++j)
        #pragma unroll
        for (int r = 0; r < 8; ++r)
          outH[(size_t)(rowb + r) * HD + j * 16 + nloc] = (bf16)x[j][r];
      return;
    } else {
      // RMSNorm over the 128-wide head (8 tiles x 16 lanes)
      #pragma unroll
      for (int r = 0; r < 8; ++r) {
        float s = 0.f;
        #pragma unroll
        for (int j = 0; j < 8; ++j) s += x[j][r] * x[j][r];
        #pragma unroll
        for (int m = 1; m < 16; m <<= 1) s += __shfl_xor(s, m, 32);
        float rstd = rsqrtf(s * (1.f / 128.f) + 1e-6f);
        #pragma unroll
        for (int j = 0; j < 8; ++j)
          x[j][r] = x[j][r] * rstd * normw[j * 16 + nloc];
      }
      // RoPE: pair (d, d+64) lives in tiles (j, j+4)
      #pragma unroll
      for (int r = 0; r < 8; ++r) {
        const int srow = rowb + r;
        #pragma unroll
        for (int j = 0; j < 4; ++j) {
          int dlo = j * 16 + nloc, dhi = dlo + 64;
          float c0 = cosT[srow * HD + dlo], s0 = sinT[srow * HD + dlo];
          float c1 = cosT[srow * HD + dhi], s1 = sinT[srow * HD + dhi];
          float lo = x[j][r] * c0 - x[j + 4][r] * s0;
          float hi = x[j + 4][r] * c1 + x[j][r] * s1;
          outH[(size_t)srow * HD + dlo] = (bf16)lo;
          outH[(size_t)srow * HD + dhi] = (bf16)hi;
        }
      }
    }
  }
}

// -------------------- flash attention (GQA, causal) --------------------
__global__ void __launch_bounds__(128, 1)
attn_kernel(const bf16* __restrict__ Qb,   // [NH][SEQ][128]
            const bf16* __restrict__ Kb,   // [NKV][SEQ][128]
            const bf16* __restrict__ Vb,   // [NKV][SEQ][128]
            bf16* __restrict__ AO)         // [SEQ][HIDDEN]
{
  constexpr int LQ = 136;                  // 128 + 8 pad (64 DW + 4 DW pad rows)
  constexpr int LV = 72;                   // 64 + 8 pad
  __shared__ bf16 Qs [64  * LQ];
  __shared__ bf16 Ks [64  * LQ];
  __shared__ bf16 Vts[128 * LV];           // V transposed: [d][k]
  __shared__ bf16 Ps [4 * 16 * LV];        // per-wave P tiles

  const int tid  = threadIdx.x, lane = tid & 31, wv = tid >> 5;
  const int qBase = blockIdx.x * 64;
  const int h = blockIdx.y, kvh = h >> 2;

  const bf16* Qh = Qb + (size_t)h   * SEQ * HD;
  const bf16* Kh = Kb + (size_t)kvh * SEQ * HD;
  const bf16* Vh = Vb + (size_t)kvh * SEQ * HD;

  // Q tile 64x128 via TDM (pad 64DW rows by 4DW -> LQ=136); waited inside loop
  if (wv == 0) {
    tdm_load_2d_bf16(lds_off(&Qs[0]), &Qh[(size_t)qBase * HD],
                     /*tile*/HD, 64, /*tensor*/HD, SEQ,
                     /*stride*/HD, /*pad_int 64DW*/5, /*pad_amt 4DW*/3, true);
  }

  const int nloc = lane & 15;
  const int rhi  = (lane >> 4) << 3;
  const int rowb = qBase + wv * 16 + rhi;

  v8f  oacc[8];
  float mrun[8], lrun[8];
  #pragma unroll
  for (int j = 0; j < 8; ++j) oacc[j] = vzero();
  #pragma unroll
  for (int r = 0; r < 8; ++r) { mrun[r] = -1e30f; lrun[r] = 0.f; }

  const float scale = 0.08838834764831845f;    // 1/sqrt(128)

  for (int kt = 0; kt <= (int)blockIdx.x; ++kt) {
    const int kBase = kt * 64;
    __syncthreads();
    // K tile 64x128 via TDM; V tile transposed manually (overlaps the DMA)
    if (wv == 0) {
      tdm_load_2d_bf16(lds_off(&Ks[0]), &Kh[(size_t)kBase * HD],
                       HD, 64, HD, SEQ, HD, 5, 3, true);
    }
    #pragma unroll
    for (int i = 0; i < 8; ++i) {
      int c = tid + i * 128;
      int k = c >> 4, d8 = (c & 15) << 3;
      v8bf v = *(const v8bf*)&Vh[(size_t)(kBase + k) * HD + d8];
      #pragma unroll
      for (int e = 0; e < 8; ++e) Vts[(d8 + e) * LV + k] = v[e];
    }
    if (wv == 0) __builtin_amdgcn_s_wait_tensorcnt(0);
    __syncthreads();

    // S = Q K^T : 16 rows x 64 keys per wave
    v8f sacc[4];
    #pragma unroll
    for (int c = 0; c < 4; ++c) sacc[c] = vzero();
    #pragma unroll
    for (int kk = 0; kk < 4; ++kk) {
      v16bf af = load_a_frag(&Qs[(wv * 16) * LQ + kk * 32], LQ, lane);
      #pragma unroll
      for (int c = 0; c < 4; ++c) {
        v16bf bfr = load_b_frag(&Ks[(c * 16) * LQ + kk * 32], LQ, lane);
        sacc[c] = wmma_bf16(af, bfr, sacc[c]);
      }
    }
    // scale + causal mask
    #pragma unroll
    for (int c = 0; c < 4; ++c) {
      int col = kBase + c * 16 + nloc;
      #pragma unroll
      for (int r = 0; r < 8; ++r) {
        float v = sacc[c][r] * scale;
        if (col > rowb + r) v = -1e30f;
        sacc[c][r] = v;
      }
    }
    // online softmax
    float sf[8];
    #pragma unroll
    for (int r = 0; r < 8; ++r) {
      float mx = fmaxf(fmaxf(sacc[0][r], sacc[1][r]), fmaxf(sacc[2][r], sacc[3][r]));
      #pragma unroll
      for (int m = 1; m < 16; m <<= 1) mx = fmaxf(mx, __shfl_xor(mx, m, 32));
      float mnew = fmaxf(mrun[r], mx);
      sf[r] = __expf(mrun[r] - mnew);
      float ls = 0.f;
      #pragma unroll
      for (int c = 0; c < 4; ++c) {
        float p = __expf(sacc[c][r] - mnew);
        sacc[c][r] = p;
        ls += p;
      }
      #pragma unroll
      for (int m = 1; m < 16; m <<= 1) ls += __shfl_xor(ls, m, 32);
      lrun[r] = lrun[r] * sf[r] + ls;
      mrun[r] = mnew;
    }
    #pragma unroll
    for (int j = 0; j < 8; ++j)
      #pragma unroll
      for (int r = 0; r < 8; ++r) oacc[j][r] *= sf[r];

    // P (C-layout) -> LDS transpose into A-fragment layout (wave-local)
    bf16* pw = &Ps[wv * 16 * LV];
    #pragma unroll
    for (int c = 0; c < 4; ++c)
      #pragma unroll
      for (int r = 0; r < 8; ++r)
        pw[(rhi + r) * LV + c * 16 + nloc] = (bf16)sacc[c][r];
    asm volatile("" ::: "memory");

    // O += P * V
    #pragma unroll
    for (int kk = 0; kk < 2; ++kk) {
      v16bf pf = load_a_frag(pw + kk * 32, LV, lane);
      #pragma unroll
      for (int j = 0; j < 8; ++j) {
        v16bf vf = load_b_frag(&Vts[(j * 16) * LV + kk * 32], LV, lane);
        oacc[j] = wmma_bf16(pf, vf, oacc[j]);
      }
    }
  }

  // normalize and store attn-out activations [SEQ][HIDDEN] (bf16)
  #pragma unroll
  for (int r = 0; r < 8; ++r) {
    float inv = 1.f / lrun[r];
    #pragma unroll
    for (int j = 0; j < 8; ++j)
      AO[(size_t)(rowb + r) * HIDDEN + h * HD + j * 16 + nloc] =
          (bf16)(oacc[j][r] * inv);
  }
}

// -------------------- launch --------------------
extern "C" void kernel_launch(void* const* d_in, const int* in_sizes, int n_in,
                              void* d_out, int out_size, void* d_ws, size_t ws_size,
                              hipStream_t stream) {
  (void)in_sizes; (void)n_in; (void)out_size; (void)ws_size;
  const float* hs   = (const float*)d_in[0];
  const float* cosT = (const float*)d_in[1];
  const float* sinT = (const float*)d_in[2];
  /* d_in[3] = attention_mask: causal, recomputed inline */
  const float* wq   = (const float*)d_in[4];
  const float* bq   = (const float*)d_in[5];
  const float* wk   = (const float*)d_in[6];
  const float* bk   = (const float*)d_in[7];
  const float* wvp  = (const float*)d_in[8];
  const float* bv   = (const float*)d_in[9];
  const float* wo   = (const float*)d_in[10];
  const float* qnw  = (const float*)d_in[11];
  const float* knw  = (const float*)d_in[12];
  float* out = (float*)d_out;

  char* ws = (char*)d_ws;
  bf16* Xb = (bf16*)ws;  ws += (size_t)SEQ * HIDDEN * 2;        // 16 MB
  bf16* Qb = (bf16*)ws;  ws += (size_t)NHEADS * SEQ * HD * 2;   // 16 MB
  bf16* Kb = (bf16*)ws;  ws += (size_t)NKVH   * SEQ * HD * 2;   //  4 MB
  bf16* Vb = (bf16*)ws;  ws += (size_t)NKVH   * SEQ * HD * 2;   //  4 MB
  bf16* AO = (bf16*)ws;                                         // 16 MB

  cvt_f32_bf16<<<(SEQ * HIDDEN / 4 + 255) / 256, 256, 0, stream>>>(hs, Xb, SEQ * HIDDEN / 4);

  dim3 blk(128);
  gemm_kernel<1><<<dim3(SEQ / 64, NHEADS), blk, 0, stream>>>(Xb, wq, bq, qnw, cosT, sinT, Qb, nullptr);
  gemm_kernel<2><<<dim3(SEQ / 64, NKVH),   blk, 0, stream>>>(Xb, wk, bk, knw, cosT, sinT, Kb, nullptr);
  gemm_kernel<3><<<dim3(SEQ / 64, NKVH),   blk, 0, stream>>>(Xb, wvp, bv, nullptr, nullptr, nullptr, Vb, nullptr);
  attn_kernel   <<<dim3(SEQ / 64, NHEADS), blk, 0, stream>>>(Qb, Kb, Vb, AO);
  gemm_kernel<0><<<dim3(SEQ / 64, HIDDEN / 128), blk, 0, stream>>>(AO, wo, nullptr, nullptr, nullptr, nullptr, nullptr, out);
}
